// TriXFFN_51934744543431
// MI455X (gfx1250) — matrix operand order repaired and checked by reading, not compiled, not run on hardware
//
#include <hip/hip_runtime.h>
#include <hip/hip_bf16.h>
#include <stdint.h>

#define D_MODEL 1024
#define D_FF    4096
#define NEXP    4
#define NTOK    16384          // 8 * 2048
#define AST     40             // LDS row stride (bf16 elems): 80B, conflict-free, 16B-aligned chunks

// ---- CDNA5 feature probes -------------------------------------------------
#if defined(__has_builtin)
#  if __has_builtin(__builtin_amdgcn_cvt_pk_bf16_f32)
#    define HAVE_CVT_PK_BF16 1
#  endif
#  if __has_builtin(__builtin_amdgcn_global_load_async_to_lds_b128)
#    define HAVE_ASYNC_LDS 1
#  endif
#  if __has_builtin(__builtin_amdgcn_s_wait_asynccnt)
#    define HAVE_WAIT_ASYNC_BUILTIN 1
#  endif
#endif

typedef __attribute__((ext_vector_type(16))) __bf16 bf16x16;
typedef __attribute__((ext_vector_type(8)))  float  f32x8;
typedef int v4i_vs __attribute__((vector_size(16)));   // async-LDS builtin param type

union FragU { bf16x16 v; uint4 q[2]; };

__device__ __forceinline__ unsigned short f2bf(float f) {
  union { float f; unsigned int u; } c; c.f = f;
  unsigned int r = c.u + 0x7FFFu + ((c.u >> 16) & 1u);   // RNE bf16
  return (unsigned short)(r >> 16);
}
#ifdef HAVE_CVT_PK_BF16
typedef __attribute__((ext_vector_type(2))) __bf16 bf16x2;
__device__ __forceinline__ unsigned int pk2(float a, float b) {
  union { bf16x2 v; unsigned int u; } c;
  c.v = __builtin_amdgcn_cvt_pk_bf16_f32(a, b);
  return c.u;
}
#else
__device__ __forceinline__ unsigned int pk2(float a, float b) {
  return (unsigned int)f2bf(a) | ((unsigned int)f2bf(b) << 16);
}
#endif

__device__ __forceinline__ void wait_async_le4() {
#ifdef HAVE_ASYNC_LDS
#  ifdef HAVE_WAIT_ASYNC_BUILTIN
  __builtin_amdgcn_s_wait_asynccnt(4);
#  else
  asm volatile("s_wait_asynccnt 0x4" ::: "memory");
#  endif
#endif
}
__device__ __forceinline__ void wait_async_le0() {
#ifdef HAVE_ASYNC_LDS
#  ifdef HAVE_WAIT_ASYNC_BUILTIN
  __builtin_amdgcn_s_wait_asynccnt(0);
#  else
  asm volatile("s_wait_asynccnt 0x0" ::: "memory");
#  endif
#endif
}

// copy 32B (one staged half-row) global(bf16) -> LDS, async DMA when available
__device__ __forceinline__ void stage32(const unsigned short* src, unsigned short* dst) {
#ifdef HAVE_ASYNC_LDS
  __builtin_amdgcn_global_load_async_to_lds_b128((v4i_vs*)src, (v4i_vs*)dst, 0, 0);
  __builtin_amdgcn_global_load_async_to_lds_b128((v4i_vs*)src, (v4i_vs*)dst, 16, 0);
#else
  ((uint4*)dst)[0] = ((const uint4*)src)[0];
  ((uint4*)dst)[1] = ((const uint4*)src)[1];
#endif
}

// ---------------- one-shot fp32 -> bf16 conversion (16 elems/thread) -------
__global__ __launch_bounds__(256) void cvt_kernel(const float* __restrict__ src,
                                                  unsigned short* __restrict__ dst) {
  size_t i = ((size_t)blockIdx.x * 256 + threadIdx.x) * 16;
  const float4* s = (const float4*)(src + i);
  float4 v0 = s[0], v1 = s[1], v2 = s[2], v3 = s[3];
  uint4 u0 = { pk2(v0.x, v0.y), pk2(v0.z, v0.w), pk2(v1.x, v1.y), pk2(v1.z, v1.w) };
  uint4 u1 = { pk2(v2.x, v2.y), pk2(v2.z, v2.w), pk2(v3.x, v3.y), pk2(v3.z, v3.w) };
  uint4* d = (uint4*)(dst + i);
  d[0] = u0; d[1] = u1;
}

// ---------------- signatures: sign(sum over d_ff rows of Wup) --------------
__global__ void sigpart_kernel(const float* __restrict__ Wup, float* __restrict__ part) {
  int t = blockIdx.x, ch = blockIdx.y, c = threadIdx.x;
  const float* p = Wup + ((size_t)t * D_FF + (size_t)ch * 256) * D_MODEL + c;
  float s = 0.f;
  #pragma unroll 8
  for (int f = 0; f < 256; ++f) s += p[(size_t)f * D_MODEL];
  part[((t * 16 + ch) * D_MODEL) + c] = s;
}
__global__ void sigfin_kernel(const float* __restrict__ part, float* __restrict__ sig) {
  int t = blockIdx.x, c = threadIdx.x;
  float s = 0.f;
  #pragma unroll
  for (int ch = 0; ch < 16; ++ch) s += part[(t * 16 + ch) * D_MODEL + c];
  sig[t * D_MODEL + c] = (s > 0.f) ? 1.f : ((s < 0.f) ? -1.f : 0.f);
}

// ---------------- routing --------------------------------------------------
__global__ __launch_bounds__(256) void route_kernel(
    const float* __restrict__ x, const float* __restrict__ sig,
    int* __restrict__ winner, int* __restrict__ counts, float* __restrict__ gate) {
  int wave = threadIdx.x >> 5, lane = threadIdx.x & 31;
  int n = blockIdx.x * 8 + wave;
  const float* xr = x + (size_t)n * D_MODEL;
  float s0 = 0.f, s1 = 0.f, s2 = 0.f, s3 = 0.f;
  for (int i = lane; i < D_MODEL; i += 32) {
    float xv = xr[i];
    s0 += xv * sig[i];
    s1 += xv * sig[D_MODEL + i];
    s2 += xv * sig[2 * D_MODEL + i];
    s3 += xv * sig[3 * D_MODEL + i];
  }
  #pragma unroll
  for (int off = 16; off > 0; off >>= 1) {
    s0 += __shfl_xor(s0, off, 32);
    s1 += __shfl_xor(s1, off, 32);
    s2 += __shfl_xor(s2, off, 32);
    s3 += __shfl_xor(s3, off, 32);
  }
  if (lane == 0) {
    int w = 0; float b = s0;                    // strict > : first-wins ties == jnp.argmax
    if (s1 > b) { b = s1; w = 1; }
    if (s2 > b) { b = s2; w = 2; }
    if (s3 > b) { b = s3; w = 3; }
    winner[n] = w;
    atomicAdd(&counts[w], 1);
    float4 g = make_float4(w == 0 ? 1.f : 0.f, w == 1 ? 1.f : 0.f,
                           w == 2 ? 1.f : 0.f, w == 3 ? 1.f : 0.f);
    *(float4*)(gate + (size_t)n * 4) = g;
  }
}

// meta (ints): [0..3]=counts [4..7]=cursor [8..12]=base [13..17]=tileBase
__global__ void scan_kernel(int* __restrict__ meta) {
  int b = 0, t = 0;
  #pragma unroll
  for (int e = 0; e < NEXP; ++e) {
    meta[8 + e]  = b;
    meta[13 + e] = t;
    b += meta[e];
    t += (meta[e] + 127) >> 7;
  }
  meta[12] = b;
  meta[17] = t;
}

__global__ __launch_bounds__(256) void scatter_kernel(
    const int* __restrict__ winner, int* __restrict__ meta, int* __restrict__ perm) {
  int n = blockIdx.x * blockDim.x + threadIdx.x;
  int w = winner[n];
  int pos = atomicAdd(&meta[4 + w], 1);
  perm[meta[8 + w] + pos] = n;
}

// ---------------- phase A: hidden = relu(xbf @ Wupbf[e].T), grouped --------
__global__ __launch_bounds__(256) void up_kernel(
    const unsigned short* __restrict__ xbf, const unsigned short* __restrict__ wupbf,
    const int* __restrict__ meta, const int* __restrict__ perm,
    unsigned short* __restrict__ hidden) {
  __shared__ int tokS[128];
  __shared__ __align__(16) unsigned short As[2][128 * AST];
  __shared__ __align__(16) unsigned short Bs[2][128 * AST];

  int gx = blockIdx.x;
  int e = -1, j = 0;
  #pragma unroll
  for (int t = 0; t < NEXP; ++t)
    if (gx >= meta[13 + t] && gx < meta[14 + t]) { e = t; j = gx - meta[13 + t]; }
  if (e < 0) return;
  int rowsValid = meta[e] - j * 128; if (rowsValid > 128) rowsValid = 128;
  int tokBase = meta[8 + e] + j * 128;

  int tid = threadIdx.x;
  if (tid < 128) tokS[tid] = perm[tokBase + (tid < rowsValid ? tid : 0)];
  __syncthreads();

  int f0 = blockIdx.y * 128;
  int wave = tid >> 5, lane = tid & 31;
  int ln = lane & 15, lh = lane >> 4;

  f32x8 acc[8];
  f32x8 zero = {0.f, 0.f, 0.f, 0.f, 0.f, 0.f, 0.f, 0.f};
  #pragma unroll
  for (int i = 0; i < 8; ++i) acc[i] = zero;

  int sr = tid >> 1;                // staged row 0..127 (two threads per row)
  int sc = (tid & 1) * 16;          // column half
  // padded rows clamp to a valid row: garbage D rows are never stored
  const unsigned short* aSrc = xbf + (size_t)tokS[sr] * D_MODEL + sc;
  const unsigned short* bSrc = wupbf + ((size_t)e * D_FF + f0 + sr) * D_MODEL + sc;
  unsigned short* aDst[2] = { &As[0][sr * AST + sc], &As[1][sr * AST + sc] };
  unsigned short* bDst[2] = { &Bs[0][sr * AST + sc], &Bs[1][sr * AST + sc] };

  auto compute = [&](const unsigned short* Ab, const unsigned short* Bb) {
    FragU b;                                 // B[k][n]: 16 consecutive K per half-wave
    b.q[0] = *(const uint4*)&Bb[(wave * 16 + ln) * AST + lh * 16];
    b.q[1] = *(const uint4*)&Bb[(wave * 16 + ln) * AST + lh * 16 + 8];
    #pragma unroll
    for (int mt = 0; mt < 8; ++mt) {
      FragU a;                               // A[m][k]: 16 consecutive K per half-wave
      a.q[0] = *(const uint4*)&Ab[(mt * 16 + ln) * AST + lh * 16];
      a.q[1] = *(const uint4*)&Ab[(mt * 16 + ln) * AST + lh * 16 + 8];
      acc[mt] = __builtin_amdgcn_wmma_f32_16x16x32_bf16(
          false, a.v, false, b.v, (short)0, acc[mt], false, false);
    }
  };

  stage32(aSrc, aDst[0]);
  stage32(bSrc, bDst[0]);

  int cur = 0;
  // steady state: unconditionally prefetch next buffer, wait<=4 (in-order drain)
  for (int k0 = 0; k0 < D_MODEL - 32; k0 += 32) {
    stage32(aSrc + k0 + 32, aDst[cur ^ 1]);
    stage32(bSrc + k0 + 32, bDst[cur ^ 1]);
    wait_async_le4();
    __syncthreads();
    compute(As[cur], Bs[cur]);
    __syncthreads();                         // all waves done reading before buffer reuse
    cur ^= 1;
  }
  // peeled tail: no prefetch, full drain
  wait_async_le0();
  __syncthreads();
  compute(As[cur], Bs[cur]);

  // ReLU + store bf16 hidden in PERMUTED row order
  int nGlob = f0 + wave * 16 + ln;
  #pragma unroll
  for (int mt = 0; mt < 8; ++mt) {
    #pragma unroll
    for (int r = 0; r < 8; ++r) {
      int m = mt * 16 + r + 8 * lh;
      if (m < rowsValid) {
        float v = acc[mt][r]; v = v > 0.f ? v : 0.f;
        hidden[(size_t)(tokBase + m) * D_FF + nGlob] = f2bf(v);
      }
    }
  }
}

// ---------------- phase B: out = hidden @ Wdownbf[e].T, scatter ------------
__global__ __launch_bounds__(256) void down_kernel(
    const unsigned short* __restrict__ hidden, const unsigned short* __restrict__ wdnbf,
    const int* __restrict__ meta, const int* __restrict__ perm,
    float* __restrict__ out) {
  __shared__ int tokS[128];
  __shared__ __align__(16) unsigned short As[2][128 * AST];
  __shared__ __align__(16) unsigned short Bs[2][128 * AST];

  int gx = blockIdx.x;
  int e = -1, j = 0;
  #pragma unroll
  for (int t = 0; t < NEXP; ++t)
    if (gx >= meta[13 + t] && gx < meta[14 + t]) { e = t; j = gx - meta[13 + t]; }
  if (e < 0) return;
  int rowsValid = meta[e] - j * 128; if (rowsValid > 128) rowsValid = 128;
  int tokBase = meta[8 + e] + j * 128;

  int tid = threadIdx.x;
  if (tid < 128) tokS[tid] = perm[tokBase + (tid < rowsValid ? tid : 0)];
  __syncthreads();

  int c0 = blockIdx.y * 128;
  int wave = tid >> 5, lane = tid & 31;
  int ln = lane & 15, lh = lane >> 4;

  f32x8 acc[8];
  f32x8 zero = {0.f, 0.f, 0.f, 0.f, 0.f, 0.f, 0.f, 0.f};
  #pragma unroll
  for (int i = 0; i < 8; ++i) acc[i] = zero;

  int sr = tid >> 1;
  int sc = (tid & 1) * 16;
  int hrowIdx = tokBase + (sr < rowsValid ? sr : 0);       // clamp pad rows
  const unsigned short* aSrc = hidden + (size_t)hrowIdx * D_FF + sc;
  const unsigned short* bSrc = wdnbf + ((size_t)e * D_MODEL + c0 + sr) * D_FF + sc;
  unsigned short* aDst[2] = { &As[0][sr * AST + sc], &As[1][sr * AST + sc] };
  unsigned short* bDst[2] = { &Bs[0][sr * AST + sc], &Bs[1][sr * AST + sc] };

  auto compute = [&](const unsigned short* Ab, const unsigned short* Bb) {
    FragU b;
    b.q[0] = *(const uint4*)&Bb[(wave * 16 + ln) * AST + lh * 16];
    b.q[1] = *(const uint4*)&Bb[(wave * 16 + ln) * AST + lh * 16 + 8];
    #pragma unroll
    for (int mt = 0; mt < 8; ++mt) {
      FragU a;
      a.q[0] = *(const uint4*)&Ab[(mt * 16 + ln) * AST + lh * 16];
      a.q[1] = *(const uint4*)&Ab[(mt * 16 + ln) * AST + lh * 16 + 8];
      acc[mt] = __builtin_amdgcn_wmma_f32_16x16x32_bf16(
          false, a.v, false, b.v, (short)0, acc[mt], false, false);
    }
  };

  stage32(aSrc, aDst[0]);
  stage32(bSrc, bDst[0]);

  int cur = 0;
  for (int k0 = 0; k0 < D_FF - 32; k0 += 32) {
    stage32(aSrc + k0 + 32, aDst[cur ^ 1]);
    stage32(bSrc + k0 + 32, bDst[cur ^ 1]);
    wait_async_le4();
    __syncthreads();
    compute(As[cur], Bs[cur]);
    __syncthreads();
    cur ^= 1;
  }
  wait_async_le0();
  __syncthreads();
  compute(As[cur], Bs[cur]);

  int cGlob = c0 + wave * 16 + ln;
  #pragma unroll
  for (int mt = 0; mt < 8; ++mt) {
    #pragma unroll
    for (int r = 0; r < 8; ++r) {
      int m = mt * 16 + r + 8 * lh;
      if (m < rowsValid) {
        int tok = tokS[m];
        out[(size_t)tok * D_MODEL + cGlob] = acc[mt][r];
      }
    }
  }
}

// ---------------- host-side orchestration ----------------------------------
extern "C" void kernel_launch(void* const* d_in, const int* in_sizes, int n_in,
                              void* d_out, int out_size, void* d_ws, size_t ws_size,
                              hipStream_t stream) {
  const float* x     = (const float*)d_in[0];
  const float* Wup   = (const float*)d_in[1];
  const float* Wdown = (const float*)d_in[2];
  float* out  = (float*)d_out;
  float* gate = out + (size_t)NTOK * D_MODEL;

  // workspace layout (bytes)
  int*   meta   = (int*)d_ws;                                    // [0,128)
  int*   winner = meta + 32;                                     // [128, 65664)
  int*   perm   = winner + NTOK;                                 // [65664, 131200)
  float* sig    = (float*)((char*)d_ws + 131200);                // 4K f32
  float* part   = (float*)((char*)d_ws + 147584);                // 64K f32
  const size_t XBF_OFF = 262144;                                 // 32 MB bf16 x
  const size_t WUP_OFF = XBF_OFF + (size_t)NTOK * D_MODEL * 2;   // 32 MB bf16 Wup
  const size_t WDN_OFF = WUP_OFF + (size_t)NEXP * D_FF * D_MODEL * 2;
  const size_t HID_OFF = WDN_OFF + (size_t)NEXP * D_MODEL * D_FF * 2;
  unsigned short* xbf    = (unsigned short*)((char*)d_ws + XBF_OFF);
  unsigned short* wupbf  = (unsigned short*)((char*)d_ws + WUP_OFF);
  unsigned short* wdnbf  = (unsigned short*)((char*)d_ws + WDN_OFF);
  unsigned short* hidden = (unsigned short*)((char*)d_ws + HID_OFF);

  (void)hipMemsetAsync(d_ws, 0, 128, stream);                    // counts + cursor

  // one-shot bf16 conversion (amortized across all GEMM tiles)
  cvt_kernel<<<dim3((NTOK * D_MODEL) / 4096), 256, 0, stream>>>(x, xbf);
  cvt_kernel<<<dim3((NEXP * D_FF * D_MODEL) / 4096), 256, 0, stream>>>(Wup, wupbf);
  cvt_kernel<<<dim3((NEXP * D_MODEL * D_FF) / 4096), 256, 0, stream>>>(Wdown, wdnbf);

  sigpart_kernel<<<dim3(NEXP, 16), dim3(D_MODEL), 0, stream>>>(Wup, part);
  sigfin_kernel<<<dim3(NEXP), dim3(D_MODEL), 0, stream>>>(part, sig);
  route_kernel<<<dim3(NTOK / 8), dim3(256), 0, stream>>>(x, sig, winner, meta, gate);
  scan_kernel<<<1, 1, 0, stream>>>(meta);
  scatter_kernel<<<dim3(NTOK / 256), dim3(256), 0, stream>>>(winner, meta, perm);

  const int maxTiles = NTOK / 128 + NEXP;                        // 132
  up_kernel<<<dim3(maxTiles, D_FF / 128), dim3(256), 0, stream>>>(xbf, wupbf, meta, perm, hidden);
  down_kernel<<<dim3(maxTiles, D_MODEL / 128), dim3(256), 0, stream>>>(hidden, wdnbf, meta, perm, out);
}
